// MultiHeadAttention_61272003445053
// MI455X (gfx1250) — compile-verified, matrix operand order checked by reading, not run on hardware
//
#include <hip/hip_runtime.h>
#include <stdint.h>

// ---------------------------------------------------------------- types
typedef __bf16 bf16_t;
typedef __attribute__((ext_vector_type(16))) __bf16 v16bf;
typedef __attribute__((ext_vector_type(8)))  float  v8f;
typedef int v4i32 __attribute__((vector_size(16)));   // builtin's 128b payload

#define DIN 1024
#define SEQ 2048
#define NB  2
#define NH  16
#define HD  64

#define AS1 __attribute__((address_space(1)))
#define AS3 __attribute__((address_space(3)))

// ---- async global->LDS staging (gfx1250 ASYNCcnt path), with fallback ----
#if __has_builtin(__builtin_amdgcn_global_load_async_to_lds_b128)
#define ASYNC_COPY16(gp, lp)                                                   \
  __builtin_amdgcn_global_load_async_to_lds_b128(                              \
      (AS1 v4i32*)(uintptr_t)(gp), (AS3 v4i32*)(uint32_t)(uintptr_t)(lp), 0, 0)
#if __has_builtin(__builtin_amdgcn_s_wait_asynccnt)
#define ASYNC_WAIT() __builtin_amdgcn_s_wait_asynccnt(0)
#else
#define ASYNC_WAIT() asm volatile("s_wait_asynccnt 0x0" ::: "memory")
#endif
#else
// synchronous fallback: plain 16B copy through VGPRs
#define ASYNC_COPY16(gp, lp) (*(uint4*)(void*)(lp) = *(const uint4*)(const void*)(gp))
#define ASYNC_WAIT() ((void)0)
#endif

static __device__ inline v16bf cvt16(const float* __restrict__ p) {
  v16bf r;
#pragma unroll
  for (int i = 0; i < 16; ++i) r[i] = (__bf16)p[i];
  return r;
}
static __device__ inline v16bf ld16(const bf16_t* __restrict__ p) {
  return *(const v16bf* __restrict__)p;
}
static __device__ inline v8f wmma_bf16(v16bf a, v16bf b, v8f c) {
  // D = A(16x32 bf16) * B(32x16 bf16) + C(16x16 f32)
  return __builtin_amdgcn_wmma_f32_16x16x32_bf16(false, a, false, b, (short)0, c,
                                                 false, false);
}

// ------------------------------------------------ kernel 0: weight prep
// dst[b][c][r] = (bf16) src[b][r][c]   (make K contiguous per output column)
__global__ void k_transpose_cvt(bf16_t* __restrict__ dst, const float* __restrict__ src,
                                int batch, int R, int C) {
  size_t idx = (size_t)blockIdx.x * blockDim.x + threadIdx.x;
  size_t total = (size_t)batch * R * C;
  if (idx >= total) return;
  int c = (int)(idx % C);
  size_t t = idx / C;
  int r = (int)(t % R);
  int b = (int)(t / R);
  dst[((size_t)b * C + c) * R + r] = (bf16_t)src[idx];
}

// ------------------------------------------- kernel 1: per-head QKV GEMM
// 8 waves/block share one head's 64x32 weight tile, double-buffered in LDS
// via async loads. Each wave computes a 16x64 output tile; A operand is
// software-pipelined one K-step ahead in registers.
__global__ void __launch_bounds__(256) k_qkv_proj(
    const float* __restrict__ X, const bf16_t* __restrict__ Wt,
    const float* __restrict__ bias, bf16_t* __restrict__ out, int v_transposed) {
  __shared__ __align__(64) bf16_t Wl[2][HD * 32];   // [n 0..63][k 0..31]
  const int tid = threadIdx.x;
  const int lane = tid & 31, wave = tid >> 5;
  const int lm = lane & 15, lh = lane >> 4;
  const int mtile = blockIdx.x * 8 + wave;
  const int h = blockIdx.y;
  const int rowbase = mtile * 16;

  const float*  xrow  = X + (size_t)(rowbase + lm) * DIN;
  const bf16_t* wbase = Wt + (size_t)h * HD * DIN;

  // cooperative 4KB tile copy: one b128 per thread
  const int wrow = tid >> 2, wcol8 = (tid & 3) * 8;

  ASYNC_COPY16(wbase + (size_t)wrow * DIN + wcol8, &Wl[0][wrow * 32 + wcol8]);

  v8f z = {};
  v8f acc[4];
#pragma unroll
  for (int t = 0; t < 4; ++t) acc[t] = z;

  v16bf a = cvt16(xrow + lh * 16);                  // A for k0 = 0
  ASYNC_WAIT();
  __syncthreads();

  for (int it = 0; it < DIN / 32; ++it) {
    const int cur = it & 1;
    const int k0n = (it + 1) * 32;
    if (k0n < DIN)                                  // stage next W tile
      ASYNC_COPY16(wbase + (size_t)wrow * DIN + k0n + wcol8,
                   &Wl[cur ^ 1][wrow * 32 + wcol8]);
    v16bf a_n = a;
    if (k0n < DIN) {
      __builtin_prefetch(xrow + k0n + lh * 16, 0, 0);
      a_n = cvt16(xrow + k0n + lh * 16);            // next A (overlaps WMMAs)
    }
#pragma unroll
    for (int t = 0; t < 4; ++t) {
      v16bf b = ld16(&Wl[cur][(t * 16 + lm) * 32 + lh * 16]);
      acc[t] = wmma_bf16(a, b, acc[t]);
    }
    a = a_n;
    ASYNC_WAIT();                                   // next buffer landed
    __syncthreads();                                // everyone done with cur
  }

  const int bb   = rowbase >> 11;        // rowbase / SEQ
  const int srow = rowbase & (SEQ - 1);
#pragma unroll
  for (int t = 0; t < 4; ++t) {
    float bval = bias[h * HD + t * 16 + lm];
#pragma unroll
    for (int r = 0; r < 8; ++r) {
      float v = acc[t][r] + bval;
      int s = srow + r + 8 * lh;
      int n = t * 16 + lm;
      size_t idx;
      if (v_transposed)   // V stored [b][h][dk][s] for contiguous PV B-operand
        idx = (((size_t)bb * NH + h) * HD + n) * SEQ + s;
      else                // Q/K stored [b][h][s][dk]
        idx = (((size_t)bb * NH + h) * SEQ + s) * HD + n;
      out[idx] = (bf16_t)v;
    }
  }
}

// --------------------------------------------- kernel 2: flash attention
// 8 waves/block = 128 query rows of one (b,h). K/V tiles (32 kv) staged in
// LDS once per block (async, double-buffered) and shared by all 8 waves.
__global__ void __launch_bounds__(256) k_attn(
    const bf16_t* __restrict__ Qp, const bf16_t* __restrict__ Kp,
    const bf16_t* __restrict__ Vt, bf16_t* __restrict__ Y) {
  __shared__ __align__(64) bf16_t Kl[2][32 * HD];   // [kv 0..31][d 0..63]
  __shared__ __align__(64) bf16_t Vl[2][HD * 32];   // [d 0..63][kv 0..31]
  __shared__ __align__(64) bf16_t Pl[8][16 * 32];   // per-wave P transpose
  const int tid = threadIdx.x;
  const int lane = tid & 31, wave = tid >> 5;
  const int lm = lane & 15, lh = lane >> 4;
  const int h = blockIdx.y, b = blockIdx.z;
  const int rowbase = blockIdx.x * 128 + wave * 16;

  const bf16_t* qbase = Qp + (((size_t)b * NH + h) * SEQ + rowbase) * HD;
  const bf16_t* kbase = Kp + (((size_t)b * NH + h) * SEQ) * HD;
  const bf16_t* vbase = Vt + (((size_t)b * NH + h) * HD) * SEQ;

  // cooperative tile copies: one b128 per thread per tile
  const int krow = tid >> 3, kcol8 = (tid & 7) * 8; // K: 32 rows x 64
  const int vrow = tid >> 2, vcol8 = (tid & 3) * 8; // V: 64 rows x 32

  ASYNC_COPY16(kbase + (size_t)krow * HD + kcol8, &Kl[0][krow * HD + kcol8]);
  ASYNC_COPY16(vbase + (size_t)vrow * SEQ + vcol8, &Vl[0][vrow * 32 + vcol8]);

  v16bf qa[2];
#pragma unroll
  for (int j = 0; j < 2; ++j)
    qa[j] = ld16(qbase + (size_t)lm * HD + j * 32 + lh * 16);

  v8f z = {};
  v8f oacc[4];
#pragma unroll
  for (int t = 0; t < 4; ++t) oacc[t] = z;
  float mrow[8], lrow[8];
#pragma unroll
  for (int r = 0; r < 8; ++r) { mrow[r] = -1e30f; lrow[r] = 0.f; }

  ASYNC_WAIT();
  __syncthreads();

  for (int it = 0; it < SEQ / 32; ++it) {
    const int cur = it & 1;
    const int kv0n = (it + 1) * 32;
    if (kv0n < SEQ) {                               // stage next K/V tiles
      ASYNC_COPY16(kbase + (size_t)(kv0n + krow) * HD + kcol8,
                   &Kl[cur ^ 1][krow * HD + kcol8]);
      ASYNC_COPY16(vbase + (size_t)vrow * SEQ + kv0n + vcol8,
                   &Vl[cur ^ 1][vrow * 32 + vcol8]);
    }
    // ----- scores: q(16x64) x k^T(64x32) from LDS, two 16x16 N-subtiles
    v8f s0 = z, s1 = z;
#pragma unroll
    for (int j = 0; j < 2; ++j) {
      v16bf kb0 = ld16(&Kl[cur][(lm)*HD + j * 32 + lh * 16]);
      v16bf kb1 = ld16(&Kl[cur][(16 + lm) * HD + j * 32 + lh * 16]);
      s0 = wmma_bf16(qa[j], kb0, s0);
      s1 = wmma_bf16(qa[j], kb1, s1);
    }
    // ----- online softmax (f32), rows split across lane halves
#pragma unroll
    for (int r = 0; r < 8; ++r) {
      float a0 = s0[r] * 0.125f;              // 1/sqrt(64)
      float a1 = s1[r] * 0.125f;
      float t = fmaxf(a0, a1);
#pragma unroll
      for (int off = 1; off < 16; off <<= 1) t = fmaxf(t, __shfl_xor(t, off, 16));
      float mnew  = fmaxf(mrow[r], t);
      float alpha = __expf(mrow[r] - mnew);
      float p0 = __expf(a0 - mnew), p1 = __expf(a1 - mnew);
      float rs = p0 + p1;
#pragma unroll
      for (int off = 1; off < 16; off <<= 1) rs += __shfl_xor(rs, off, 16);
      lrow[r] = lrow[r] * alpha + rs;
      mrow[r] = mnew;
#pragma unroll
      for (int t4 = 0; t4 < 4; ++t4) oacc[t4][r] *= alpha;
      Pl[wave][(r + 8 * lh) * 32 + lm]      = (bf16_t)p0;
      Pl[wave][(r + 8 * lh) * 32 + 16 + lm] = (bf16_t)p1;
    }
    asm volatile("s_wait_dscnt 0" ::: "memory");        // cross-lane LDS RAW
    v16bf pa = ld16(&Pl[wave][lm * 32 + lh * 16]);      // A-layout reload of P
#pragma unroll
    for (int t4 = 0; t4 < 4; ++t4) {                    // O += P(16x32) x V(32x16)
      v16bf vb = ld16(&Vl[cur][(t4 * 16 + lm) * 32 + lh * 16]);
      oacc[t4] = wmma_bf16(pa, vb, oacc[t4]);
    }
    ASYNC_WAIT();                                       // next tiles landed
    __syncthreads();                                    // all waves done w/ cur
  }

  // normalize and write Y[b][s][h*64+n] (bf16) for the output GEMM
#pragma unroll
  for (int r = 0; r < 8; ++r) {
    float inv = 1.f / lrow[r];
    int s = rowbase + r + 8 * lh;
#pragma unroll
    for (int t4 = 0; t4 < 4; ++t4) {
      int n = h * HD + t4 * 16 + lm;
      Y[((size_t)b * SEQ + s) * (NH * HD) + n] = (bf16_t)(oacc[t4][r] * inv);
    }
  }
}

// ---------------------------------------------- kernel 3: output GEMM
// Same structure as k_qkv_proj: shared Wo tile double-buffered in LDS.
__global__ void __launch_bounds__(256) k_oproj(
    const bf16_t* __restrict__ Y, const bf16_t* __restrict__ Wot,
    const float* __restrict__ bo, float* __restrict__ out) {
  __shared__ __align__(64) bf16_t Wl[2][64 * 32];   // [n local 0..63][k 0..31]
  const int tid = threadIdx.x;
  const int lane = tid & 31, wave = tid >> 5;
  const int lm = lane & 15, lh = lane >> 4;
  const int mtile = blockIdx.x * 8 + wave;
  const int ntile = blockIdx.y;            // 64-wide column tile
  const int rowbase = mtile * 16;

  const bf16_t* yrow  = Y + (size_t)(rowbase + lm) * DIN;
  const bf16_t* wbase = Wot + (size_t)(ntile * 64) * DIN;

  const int wrow = tid >> 2, wcol8 = (tid & 3) * 8;
  ASYNC_COPY16(wbase + (size_t)wrow * DIN + wcol8, &Wl[0][wrow * 32 + wcol8]);

  v8f z = {};
  v8f acc[4];
#pragma unroll
  for (int t = 0; t < 4; ++t) acc[t] = z;

  v16bf a = ld16(yrow + lh * 16);
  ASYNC_WAIT();
  __syncthreads();

  for (int it = 0; it < DIN / 32; ++it) {
    const int cur = it & 1;
    const int k0n = (it + 1) * 32;
    if (k0n < DIN)
      ASYNC_COPY16(wbase + (size_t)wrow * DIN + k0n + wcol8,
                   &Wl[cur ^ 1][wrow * 32 + wcol8]);
    v16bf a_n = a;
    if (k0n < DIN) {
      __builtin_prefetch(yrow + k0n + lh * 16, 0, 0);
      a_n = ld16(yrow + k0n + lh * 16);
    }
#pragma unroll
    for (int t = 0; t < 4; ++t) {
      v16bf bm = ld16(&Wl[cur][(t * 16 + lm) * 32 + lh * 16]);
      acc[t] = wmma_bf16(a, bm, acc[t]);
    }
    a = a_n;
    ASYNC_WAIT();
    __syncthreads();
  }
#pragma unroll
  for (int t = 0; t < 4; ++t) {
    int n = ntile * 64 + t * 16 + lm;
    float bval = bo[n];
#pragma unroll
    for (int r = 0; r < 8; ++r) {
      int g = rowbase + r + 8 * lh;
      out[(size_t)g * DIN + n] = acc[t][r] + bval;
    }
  }
}

// ------------------------------------------------------------- launcher
extern "C" void kernel_launch(void* const* d_in, const int* in_sizes, int n_in,
                              void* d_out, int out_size, void* d_ws, size_t ws_size,
                              hipStream_t stream) {
  (void)in_sizes; (void)n_in; (void)out_size; (void)ws_size;
  const float* Q  = (const float*)d_in[0];
  const float* K  = (const float*)d_in[1];
  const float* V  = (const float*)d_in[2];
  const float* Wq = (const float*)d_in[3];
  const float* bq = (const float*)d_in[4];
  const float* Wk = (const float*)d_in[5];
  const float* bk = (const float*)d_in[6];
  const float* Wv = (const float*)d_in[7];
  const float* bv = (const float*)d_in[8];
  const float* Wo = (const float*)d_in[9];
  const float* bo = (const float*)d_in[10];
  float* out = (float*)d_out;

  const size_t U = (size_t)NH * DIN * HD;   // 1,048,576 elements
  bf16_t* p   = (bf16_t*)d_ws;
  bf16_t* wqt = p;
  bf16_t* wkt = wqt + U;
  bf16_t* wvt = wkt + U;
  bf16_t* wot = wvt + U;                    // DIN*DIN == U
  bf16_t* qp  = wot + U;
  bf16_t* kp  = qp + 4 * U;
  bf16_t* vt  = kp + 4 * U;
  bf16_t* y   = vt + 4 * U;                 // total: 20*U bf16 = 40 MB

  {
    int tot = NH * DIN * HD;
    k_transpose_cvt<<<(tot + 255) / 256, 256, 0, stream>>>(wqt, Wq, NH, DIN, HD);
    k_transpose_cvt<<<(tot + 255) / 256, 256, 0, stream>>>(wkt, Wk, NH, DIN, HD);
    k_transpose_cvt<<<(tot + 255) / 256, 256, 0, stream>>>(wvt, Wv, NH, DIN, HD);
    int tot2 = DIN * DIN;
    k_transpose_cvt<<<(tot2 + 255) / 256, 256, 0, stream>>>(wot, Wo, 1, DIN, DIN);
  }

  dim3 gproj(NB * SEQ / 16 / 8, NH);        // (32, 16), 8 waves/block
  k_qkv_proj<<<gproj, 256, 0, stream>>>(Q, wqt, bq, qp, 0);
  k_qkv_proj<<<gproj, 256, 0, stream>>>(K, wkt, bk, kp, 0);
  k_qkv_proj<<<gproj, 256, 0, stream>>>(V, wvt, bv, vt, 1);

  dim3 gattn(SEQ / 128, NH, NB);            // (16, 16, 2), 8 waves/block
  k_attn<<<gattn, 256, 0, stream>>>(qp, kp, vt, y);

  dim3 gout(NB * SEQ / 16 / 8, DIN / 64);   // (32, 16)
  k_oproj<<<gout, 256, 0, stream>>>(y, wot, bo, out);
}
